// NeighborhoodEncoder_56495999811731
// MI455X (gfx1250) — compile-verified, matrix operand order checked by reading, not run on hardware
//
#include <hip/hip_runtime.h>
#include <math.h>

typedef __bf16 bf16;
typedef __attribute__((ext_vector_type(16))) __bf16 v16bf;
typedef __attribute__((ext_vector_type(8)))  float  v8f;

// Build a 16-element bf16 WMMA fragment from two contiguous 8-element runs.
// A-matrix 16x32 bf16 layout (wave32): lane L(0-15) = row M=L, holds K = kb..kb+7
// and kb+16..kb+23 where kb = (lane>=16 ? 8 : 0); B mirrors with N in place of M.
__device__ inline v16bf make_frag(const bf16* p0, const bf16* p1) {
  v16bf r;
#pragma unroll
  for (int j = 0; j < 8; ++j) { r[j] = p0[j]; r[j + 8] = p1[j]; }
  return r;
}

// Wave-level (16 x K) * (K x NOUT) -> 16 x NOUT GEMM, f32 accumulate.
// As: LDS, row-major 16 x K bf16 (wave-private). BT: global bf16, transposed
// weights NOUT x K (so each lane's B column is K-contiguous -> b128 loads).
// Cs: LDS f32, row-major 16 x LDC; writes columns [0, NOUT).
template <int K, int NOUT, int LDC>
__device__ inline void wave_gemm(const bf16* As, const bf16* BT, float* Cs, int lane) {
  const int r15  = lane & 15;
  const int koff = (lane >> 4) << 3;   // second half-wave covers K+8 / K+24
  const bf16* arow = As + r15 * K + koff;
  for (int n0 = 0; n0 < NOUT; n0 += 16) {
    const bf16* bcol = BT + (size_t)(n0 + r15) * K + koff;
    v8f acc = {};
#pragma unroll
    for (int k0 = 0; k0 < K; k0 += 32) {
      v16bf a = make_frag(arow + k0, arow + k0 + 16);
      v16bf b = make_frag(bcol + k0, bcol + k0 + 16);
      acc = __builtin_amdgcn_wmma_f32_16x16x32_bf16(false, a, false, b,
                                                    (short)0, acc, false, false);
    }
    // C/D layout: VGPR j -> row j (lanes 0-15) or row j+8 (lanes 16-31), col = n0 + (lane&15)
    const int rbase = (lane >> 4) << 3;
#pragma unroll
    for (int j = 0; j < 8; ++j) Cs[(rbase + j) * LDC + n0 + r15] = acc[j];
  }
}

// Convert + transpose weights: W[K][N] f32 -> WT[N][K] bf16 (stays L2-resident).
__global__ void convT_kernel(const float* __restrict__ W, bf16* __restrict__ WT,
                             int K, int N) {
  int i = blockIdx.x * blockDim.x + threadIdx.x;
  if (i < K * N) {
    int k = i / N, n = i - k * N;
    WT[(size_t)n * K + k] = (bf16)W[i];
  }
}

// Fused edge stage: gather x[src] * (1 + coef*strength) -> bf16 LDS tile ->
// WMMA GEMM with pool_w -> +bias, LayerNorm, ReLU -> atomic segment-max into agg.
// One wave per 16 edges; all LDS wave-private (no barriers; LDS is in-order per wave).
__global__ void __launch_bounds__(128) edge_kernel(
    const float* __restrict__ x, const int* __restrict__ eidx,
    const float* __restrict__ estr, const float* __restrict__ ewmc,
    const bf16* __restrict__ poolT, const float* __restrict__ pool_b,
    const float* __restrict__ lnp_g, const float* __restrict__ lnp_b,
    float* __restrict__ agg, int E, int NN) {
  __shared__ bf16  As[4][16 * 128];
  __shared__ float Cs[4][16 * 128];
  const int lane  = threadIdx.x & 31;
  const int w     = threadIdx.x >> 5;
  const int ebase = (blockIdx.x * 4 + w) * 16;
  const float coef = log1pf(__expf(ewmc[0]));   // softplus of the scalar

  int srcL = 0, dstL = -1;
  float scL = 0.f;
  if (lane < 16) {
    int e = ebase + lane;
    if (e < E) {
      srcL = eidx[e];
      dstL = eidx[E + e];
      scL  = 1.f + coef * estr[e];
    }
  }
  bf16* Aw = As[w];
#pragma unroll 4
  for (int m = 0; m < 16; ++m) {
    int   srow = __shfl(srcL, m);
    float sc   = __shfl(scL, m);
    const float4 xv = *reinterpret_cast<const float4*>(x + (size_t)srow * 128 + lane * 4);
    bf16* dp = Aw + m * 128 + lane * 4;
    dp[0] = (bf16)(xv.x * sc);
    dp[1] = (bf16)(xv.y * sc);
    dp[2] = (bf16)(xv.z * sc);
    dp[3] = (bf16)(xv.w * sc);
  }
  float* Cw = Cs[w];
  wave_gemm<128, 128, 128>(Aw, poolT, Cw, lane);

  // LayerNorm over 128 cols: two lanes per row, paired via shfl_xor(16).
  const int row  = lane & 15;
  const int half = lane >> 4;
  float* crow = Cw + row * 128;
  float s = 0.f, sq = 0.f;
  for (int i = 0; i < 64; ++i) {
    int c = half * 64 + i;
    float v = crow[c] + pool_b[c];
    crow[c] = v;
    s += v; sq += v * v;
  }
  s  += __shfl_xor(s, 16);
  sq += __shfl_xor(sq, 16);
  const float mean = s * (1.f / 128.f);
  const float inv  = rsqrtf(sq * (1.f / 128.f) - mean * mean + 1e-5f);
  const int d = __shfl(dstL, row);
  if (d >= 0) {
    // ReLU output >= 0 => int-bit atomicMax == float max; agg pre-zeroed
    // also provides the reference's maximum(agg, 0).
    int* ai = reinterpret_cast<int*>(agg) + (size_t)d * 128;
    for (int i = 0; i < 64; ++i) {
      int c = half * 64 + i;
      float v = fmaxf((crow[c] - mean) * inv * lnp_g[c] + lnp_b[c], 0.f);
      atomicMax(&ai[c], __float_as_int(v));
    }
  }
}

// Dense MLP stage: out = relu(LN(in @ W + b)) written as bf16 for the next GEMM.
template <int K, int NOUT, bool IN_F32>
__global__ void __launch_bounds__(64) mlp_kernel(
    const void* __restrict__ inp, const bf16* __restrict__ BT,
    const float* __restrict__ lin_b, const float* __restrict__ g,
    const float* __restrict__ beta, bf16* __restrict__ outp, int NN) {
  __shared__ bf16  As[2][16 * K];
  __shared__ float Cs[2][16 * NOUT];
  const int lane  = threadIdx.x & 31;
  const int w     = threadIdx.x >> 5;
  const int rbase = (blockIdx.x * 2 + w) * 16;
  bf16* Aw = As[w];
  for (int m = 0; m < 16; ++m) {
    int r  = rbase + m;
    int rr = (r < NN) ? r : 0;
    if (IN_F32) {
      const float* xr = (const float*)inp + (size_t)rr * K;
#pragma unroll
      for (int t = 0; t < K; t += 128) {
        float4 xv = *reinterpret_cast<const float4*>(xr + t + lane * 4);
        bf16* dp = Aw + m * K + t + lane * 4;
        dp[0] = (bf16)xv.x; dp[1] = (bf16)xv.y; dp[2] = (bf16)xv.z; dp[3] = (bf16)xv.w;
      }
    } else {
      const bf16* xr = (const bf16*)inp + (size_t)rr * K;
#pragma unroll
      for (int t = 0; t < K; t += 256) {
        const bf16* sp = xr + t + lane * 8;
        bf16* dp = Aw + m * K + t + lane * 8;
#pragma unroll
        for (int j = 0; j < 8; ++j) dp[j] = sp[j];
      }
    }
  }
  float* Cw = Cs[w];
  wave_gemm<K, NOUT, NOUT>(Aw, BT, Cw, lane);

  const int row  = lane & 15;
  const int half = lane >> 4;
  const int r    = rbase + row;
  float* crow = Cw + row * NOUT;
  const int HB = NOUT / 2;
  float s = 0.f, sq = 0.f;
  for (int i = 0; i < HB; ++i) {
    int c = half * HB + i;
    float v = crow[c] + lin_b[c];
    crow[c] = v;
    s += v; sq += v * v;
  }
  s  += __shfl_xor(s, 16);
  sq += __shfl_xor(sq, 16);
  const float mean = s / (float)NOUT;
  const float inv  = rsqrtf(sq / (float)NOUT - mean * mean + 1e-5f);
  if (r < NN) {
    for (int i = 0; i < HB; ++i) {
      int c = half * HB + i;
      float v = fmaxf((crow[c] - mean) * inv * g[c] + beta[c], 0.f);
      outp[(size_t)r * NOUT + c] = (bf16)v;
    }
  }
}

// Head: mu = h@mu_w + mu_b ; std = exp(0.5*(h@lv_w + lv_b)). Output f32,
// concatenated [mu (N*64) | std (N*64)].
__global__ void __launch_bounds__(64) head_kernel(
    const bf16* __restrict__ h, const bf16* __restrict__ muT,
    const float* __restrict__ mu_b, const bf16* __restrict__ lvT,
    const float* __restrict__ lv_b, float* __restrict__ out, int NN) {
  __shared__ bf16  As[2][16 * 256];
  __shared__ float Cs[2][16 * 128];
  const int lane  = threadIdx.x & 31;
  const int w     = threadIdx.x >> 5;
  const int rbase = (blockIdx.x * 2 + w) * 16;
  bf16* Aw = As[w];
  for (int m = 0; m < 16; ++m) {
    int r  = rbase + m;
    int rr = (r < NN) ? r : 0;
    const bf16* sp = h + (size_t)rr * 256 + lane * 8;
    bf16* dp = Aw + m * 256 + lane * 8;
#pragma unroll
    for (int j = 0; j < 8; ++j) dp[j] = sp[j];
  }
  float* Cw = Cs[w];
  wave_gemm<256, 64, 128>(Aw, muT, Cw, lane);
  wave_gemm<256, 64, 128>(Aw, lvT, Cw + 64, lane);

  const int row  = lane & 15;
  const int half = lane >> 4;
  const int r    = rbase + row;
  if (r < NN) {
    const float* crow = Cw + row * 128;
    if (half == 0) {
      for (int c = 0; c < 64; ++c)
        out[(size_t)r * 64 + c] = crow[c] + mu_b[c];
    } else {
      float* so = out + (size_t)NN * 64;
      for (int c = 0; c < 64; ++c)
        so[(size_t)r * 64 + c] = __expf(0.5f * (crow[64 + c] + lv_b[c]));
    }
  }
}

extern "C" void kernel_launch(void* const* d_in, const int* in_sizes, int n_in,
                              void* d_out, int out_size, void* d_ws, size_t ws_size,
                              hipStream_t stream) {
  const int D = 128, H1 = 256, H2 = 256, OUT = 64;
  const int NN = in_sizes[0] / D;
  const int E  = in_sizes[2];

  const float* x      = (const float*)d_in[0];
  const int*   ei     = (const int*)d_in[1];
  const float* es     = (const float*)d_in[2];
  const float* ewmc   = (const float*)d_in[3];
  const float* pool_w = (const float*)d_in[4];
  const float* pool_b = (const float*)d_in[5];
  const float* lnp_g  = (const float*)d_in[6];
  const float* lnp_b  = (const float*)d_in[7];
  const float* w1     = (const float*)d_in[8];
  const float* b1     = (const float*)d_in[9];
  const float* ln1_g  = (const float*)d_in[10];
  const float* ln1_b  = (const float*)d_in[11];
  const float* w2     = (const float*)d_in[12];
  const float* b2     = (const float*)d_in[13];
  const float* ln2_g  = (const float*)d_in[14];
  const float* ln2_b  = (const float*)d_in[15];
  const float* mu_w   = (const float*)d_in[16];
  const float* mu_b   = (const float*)d_in[17];
  const float* lv_w   = (const float*)d_in[18];
  const float* lv_b   = (const float*)d_in[19];

  char* ws = (char*)d_ws;
  size_t o = 0;
  float* agg  = (float*)(ws + o); o += (size_t)NN * D  * sizeof(float);
  bf16* h1    = (bf16*)(ws + o);  o += (size_t)NN * H1 * sizeof(bf16);
  bf16* h2    = (bf16*)(ws + o);  o += (size_t)NN * H2 * sizeof(bf16);
  bf16* poolT = (bf16*)(ws + o);  o += (size_t)D  * D  * sizeof(bf16);
  bf16* w1T   = (bf16*)(ws + o);  o += (size_t)H1 * D  * sizeof(bf16);
  bf16* w2T   = (bf16*)(ws + o);  o += (size_t)H2 * H1 * sizeof(bf16);
  bf16* muT   = (bf16*)(ws + o);  o += (size_t)OUT * H2 * sizeof(bf16);
  bf16* lvT   = (bf16*)(ws + o);  o += (size_t)OUT * H2 * sizeof(bf16);

  hipMemsetAsync(agg, 0, (size_t)NN * D * sizeof(float), stream);

  auto cg = [](int n) { return (n + 255) / 256; };
  convT_kernel<<<cg(D * D),   256, 0, stream>>>(pool_w, poolT, D,  D);
  convT_kernel<<<cg(D * H1),  256, 0, stream>>>(w1,     w1T,   D,  H1);
  convT_kernel<<<cg(H1 * H2), 256, 0, stream>>>(w2,     w2T,   H1, H2);
  convT_kernel<<<cg(H2 * OUT),256, 0, stream>>>(mu_w,   muT,   H2, OUT);
  convT_kernel<<<cg(H2 * OUT),256, 0, stream>>>(lv_w,   lvT,   H2, OUT);

  edge_kernel<<<(E + 63) / 64, 128, 0, stream>>>(x, ei, es, ewmc, poolT, pool_b,
                                                 lnp_g, lnp_b, agg, E, NN);

  const int nblk = (NN + 31) / 32;
  mlp_kernel<128, 256, true ><<<nblk, 64, 0, stream>>>(agg, w1T, b1, ln1_g, ln1_b, h1, NN);
  mlp_kernel<256, 256, false><<<nblk, 64, 0, stream>>>(h1,  w2T, b2, ln2_g, ln2_b, h2, NN);
  head_kernel<<<nblk, 64, 0, stream>>>(h2, muT, mu_b, lvT, lv_b, (float*)d_out, NN);
}
